// SocialTNN_13280038879560
// MI455X (gfx1250) — matrix operand order repaired and checked by reading, not compiled, not run on hardware
//
#include <hip/hip_runtime.h>

typedef __attribute__((ext_vector_type(16))) _Float16 v16h;
typedef __attribute__((ext_vector_type(8)))  _Float16 v8h;
typedef __attribute__((ext_vector_type(8)))  float    v8f;

struct Params {
  const int*   symbol;
  const float* other_hidden;
  const float* prev_hidden;
  const float* emb;
  const float* mem_gate;
  const float *Wm,*bm,*Wq,*bq,*Wk,*bk,*Wv,*bv;
  const float *W1,*b1,*W2,*b2,*W3,*b3,*W4,*b4;
  const float *Wa,*ba,*Ws,*bs,*Wi,*bi;
  float* out;
  int Btot;
};

// Hardware tanh: gfx1250 V_TANH_F32 (TRANS32 op). Builtin if available, else
// inline asm with an explicit TRANS32 hazard delay, per ISA §7.4/§8.4.
__device__ __forceinline__ float fast_tanh(float x) {
#if __has_builtin(__builtin_amdgcn_tanhf)
  return __builtin_amdgcn_tanhf(x);
#else
  float r;
  asm volatile("v_tanh_f32 %0, %1\n\t"
               "s_delay_alu instid0(TRANS32_DEP_1)"
               : "=v"(r) : "v"(x));
  return r;
#endif
}

// A-fragment (16x32 f16, MxK): lane L holds row m=L%16; khalf=L/16.
// v16h elems 0..7  -> K = khalf*8 + 0..7       (16B contiguous)
//       elems 8..15 -> K = 16 + khalf*8 + 0..7 (16B contiguous)
__device__ __forceinline__ v16h lds_a(const _Float16* row, int kh) {
  v8h lo = *(const v8h*)(row + kh * 8);
  v8h hi = *(const v8h*)(row + 16 + kh * 8);
  return __builtin_shufflevector(lo, hi, 0,1,2,3,4,5,6,7,8,9,10,11,12,13,14,15);
}
// B-fragment (32x16 f16, KxN) with B stored transposed (N-major, 32 halves/row):
// lane L holds col n=L%16; khalf=L/16; elems 0..7 -> K = khalf*8 + 0..7, elems 8..15 -> K = 16 + khalf*8 + 0..7
__device__ __forceinline__ v16h lds_b(const _Float16* row, int kh) {
  v8h lo = *(const v8h*)(row + kh * 8);
  v8h hi = *(const v8h*)(row + 16 + kh * 8);
  return __builtin_shufflevector(lo, hi, 0,1,2,3,4,5,6,7,8,9,10,11,12,13,14,15);
}
__device__ __forceinline__ v8f wmma_f16(v16h a, v16h b, v8f c) {
  return __builtin_amdgcn_wmma_f32_16x16x32_f16(false, a, false, b, (short)0, c, false, false);
}

__global__ __launch_bounds__(128) void social_tnn_kernel(Params p) {
  // ---- LDS (all B-operands stored N-major so B-frags are contiguous) ----
  __shared__ __attribute__((aligned(16))) _Float16 wtAll[8][32][32]; // Wm,Wq,Wk,Wv,W1..W4 transposed [n][k]
  __shared__ __attribute__((aligned(16))) _Float16 wsT[256][32];     // Ws transposed
  __shared__ __attribute__((aligned(16))) _Float16 kMat[64][32];     // k (neighbor-major = Bt for scores)
  __shared__ __attribute__((aligned(16))) _Float16 vTr[32][64];      // v transposed [hidden][neighbor]
  __shared__ __attribute__((aligned(16))) _Float16 hTile[4][16][32]; // per-wave h
  __shared__ __attribute__((aligned(16))) _Float16 qTile[4][16][32]; // per-wave q
  __shared__ __attribute__((aligned(16))) _Float16 wTile[4][16][64]; // per-wave attn weights
  __shared__ float biasM[32], biasQ[32], biasK[32], biasV[32];
  __shared__ float biasMlp[4][32];
  __shared__ float biasS[256];
  __shared__ float gateL[32], WaL[32], WiL[96], biL[3], baL[1];
  __shared__ int   symL[4][16];

  const int tid  = threadIdx.x;
  const int lane = tid & 31;
  const int w    = tid >> 5;   // wave id 0..3, each owns 16 agents
  const int nl   = lane & 15;
  const int kh   = lane >> 4;
  const v8f vzero = {0.f,0.f,0.f,0.f,0.f,0.f,0.f,0.f};

  // ---- phase 0: stage weights/biases (f32 -> transposed f16) ----
  {
    const float* Ms[8] = {p.Wm, p.Wq, p.Wk, p.Wv, p.W1, p.W2, p.W3, p.W4};
    for (int mat = 0; mat < 8; ++mat) {
      const float* W = Ms[mat];
      for (int i = tid; i < 32 * 32; i += 128) {
        int n = i >> 5, k = i & 31;
        wtAll[mat][n][k] = (_Float16)W[k * 32 + n];
      }
    }
  }
  for (int i = tid; i < 256 * 32; i += 128) {
    int n = i >> 5, k = i & 31;
    wsT[n][k] = (_Float16)p.Ws[k * 256 + n];
  }
  // other_hidden staged f16 into wTile scratch (wTile not live until softmax)
  _Float16 (* const ohT)[32] = (_Float16(*)[32])&wTile[0][0][0];
  for (int i = tid; i < 64 * 32; i += 128) ohT[i >> 5][i & 31] = (_Float16)p.other_hidden[i];
  if (tid < 32) {
    biasM[tid] = p.bm[tid]; biasQ[tid] = p.bq[tid];
    biasK[tid] = p.bk[tid]; biasV[tid] = p.bv[tid];
    biasMlp[0][tid] = p.b1[tid]; biasMlp[1][tid] = p.b2[tid];
    biasMlp[2][tid] = p.b3[tid]; biasMlp[3][tid] = p.b4[tid];
    gateL[tid] = 1.f / (1.f + __expf(-p.mem_gate[tid]));
    WaL[tid]   = p.Wa[tid];
  }
  for (int i = tid; i < 256; i += 128) biasS[i] = p.bs[i];
  if (tid < 96) WiL[tid] = p.Wi[tid];
  if (tid < 3)  biL[tid] = p.bi[tid];
  if (tid == 0) baL[0]   = p.ba[0];
  __syncthreads();

  // ---- phase 1: k = oh@Wk+bk, v = oh@Wv+bv (wave w handles neighbor rows 16w..16w+15) ----
  {
    v16h a = lds_a(&ohT[w * 16 + nl][0], kh);
    #pragma unroll
    for (int t = 0; t < 2; ++t) {
      v16h bk = lds_b(&wtAll[2][t * 16 + nl][0], kh);
      v16h bv = lds_b(&wtAll[3][t * 16 + nl][0], kh);
      v8f ck = wmma_f16(a, bk, vzero);
      v8f cv = wmma_f16(a, bv, vzero);
      int n = t * 16 + nl;
      #pragma unroll
      for (int g = 0; g < 8; ++g) {
        int m = w * 16 + g + 8 * kh;
        kMat[m][n] = (_Float16)(ck[g] + biasK[n]);
        vTr[n][m]  = (_Float16)(cv[g] + biasV[n]);
      }
    }
  }
  __syncthreads();

  const long long base = (long long)blockIdx.x * 64 + w * 16;
  if (lane < 16) symL[w][lane] = p.symbol[base + lane];
  __syncthreads();

  // ---- h0 = emb*(1-gate) + tanh(prev@Wm+bm)*gate ----
  {
    const float* prow = p.prev_hidden + (base + nl) * 32;
    const float4* p4 = (const float4*)prow;
    float4 f0 = p4[kh * 2 + 0], f1 = p4[kh * 2 + 1];
    float4 f2 = p4[kh * 2 + 4], f3 = p4[kh * 2 + 5];
    v16h a;
    a[0]=(_Float16)f0.x; a[1]=(_Float16)f0.y; a[2]=(_Float16)f0.z; a[3]=(_Float16)f0.w;
    a[4]=(_Float16)f1.x; a[5]=(_Float16)f1.y; a[6]=(_Float16)f1.z; a[7]=(_Float16)f1.w;
    a[8]=(_Float16)f2.x; a[9]=(_Float16)f2.y; a[10]=(_Float16)f2.z; a[11]=(_Float16)f2.w;
    a[12]=(_Float16)f3.x; a[13]=(_Float16)f3.y; a[14]=(_Float16)f3.z; a[15]=(_Float16)f3.w;
    #pragma unroll
    for (int t = 0; t < 2; ++t) {
      v16h b = lds_b(&wtAll[0][t * 16 + nl][0], kh);
      v8f c = wmma_f16(a, b, vzero);
      int n = t * 16 + nl;
      float gt = gateL[n];
      #pragma unroll
      for (int g = 0; g < 8; ++g) {
        int m = g + 8 * kh;
        float mem = fast_tanh(c[g] + biasM[n]);
        float ev  = p.emb[symL[w][m] * 32 + n];
        hTile[w][m][n] = (_Float16)(ev * (1.f - gt) + mem * gt);
      }
    }
  }
  __syncthreads();

  // ---- q = h@Wq + bq ----
  {
    v16h a = lds_a(&hTile[w][nl][0], kh);
    #pragma unroll
    for (int t = 0; t < 2; ++t) {
      v16h b = lds_b(&wtAll[1][t * 16 + nl][0], kh);
      v8f c = wmma_f16(a, b, vzero);
      int n = t * 16 + nl;
      #pragma unroll
      for (int g = 0; g < 8; ++g) qTile[w][g + 8 * kh][n] = (_Float16)(c[g] + biasQ[n]);
    }
  }
  __syncthreads();

  // ---- scores = q@k.T / sqrt(32); softmax in registers on D layout ----
  {
    v16h a = lds_a(&qTile[w][nl][0], kh);
    v8f s[4];
    #pragma unroll
    for (int t = 0; t < 4; ++t) s[t] = wmma_f16(a, lds_b(&kMat[t * 16 + nl][0], kh), vzero);
    const float inv = 0.17677669529663687f; // 1/sqrt(32)
    #pragma unroll
    for (int g = 0; g < 8; ++g) {
      float x0 = s[0][g]*inv, x1 = s[1][g]*inv, x2 = s[2][g]*inv, x3 = s[3][g]*inv;
      float mx = fmaxf(fmaxf(x0, x1), fmaxf(x2, x3));
      mx = fmaxf(mx, __shfl_xor(mx, 1, 32));
      mx = fmaxf(mx, __shfl_xor(mx, 2, 32));
      mx = fmaxf(mx, __shfl_xor(mx, 4, 32));
      mx = fmaxf(mx, __shfl_xor(mx, 8, 32));
      float e0 = __expf(x0 - mx), e1 = __expf(x1 - mx), e2 = __expf(x2 - mx), e3 = __expf(x3 - mx);
      float sm = e0 + e1 + e2 + e3;
      sm += __shfl_xor(sm, 1, 32);
      sm += __shfl_xor(sm, 2, 32);
      sm += __shfl_xor(sm, 4, 32);
      sm += __shfl_xor(sm, 8, 32);
      float r = 1.f / sm;
      int m = g + 8 * kh;
      wTile[w][m][ 0 + nl] = (_Float16)(e0 * r);
      wTile[w][m][16 + nl] = (_Float16)(e1 * r);
      wTile[w][m][32 + nl] = (_Float16)(e2 * r);
      wTile[w][m][48 + nl] = (_Float16)(e3 * r);
    }
  }
  __syncthreads();

  // ---- h += (weights@v) * 0.3  (K=64 -> two accumulated WMMAs per n-tile) ----
  {
    v16h a0 = lds_a(&wTile[w][nl][0], kh);
    v16h a1 = lds_a(&wTile[w][nl][32], kh);
    #pragma unroll
    for (int t = 0; t < 2; ++t) {
      v16h b0 = lds_b(&vTr[t * 16 + nl][0], kh);
      v16h b1 = lds_b(&vTr[t * 16 + nl][32], kh);
      v8f c = wmma_f16(a0, b0, vzero);
      c = wmma_f16(a1, b1, c);
      int n = t * 16 + nl;
      #pragma unroll
      for (int g = 0; g < 8; ++g) {
        int m = g + 8 * kh;
        hTile[w][m][n] = (_Float16)((float)hTile[w][m][n] + 0.3f * c[g]);
      }
    }
  }
  __syncthreads();

  // ---- 4x gated-residual tanh MLP blocks ----
  for (int j = 0; j < 4; ++j) {
    v16h a = lds_a(&hTile[w][nl][0], kh);
    v8f c0 = wmma_f16(a, lds_b(&wtAll[4 + j][ 0 + nl][0], kh), vzero);
    v8f c1 = wmma_f16(a, lds_b(&wtAll[4 + j][16 + nl][0], kh), vzero);
    #pragma unroll
    for (int t = 0; t < 2; ++t) {
      int n = t * 16 + nl;
      float bj = biasMlp[j][n];
      #pragma unroll
      for (int g = 0; g < 8; ++g) {
        int m = g + 8 * kh;
        float acc = (t == 0) ? c0[g] : c1[g];
        hTile[w][m][n] = (_Float16)((float)hTile[w][m][n] + 0.3f * fast_tanh(acc + bj));
      }
    }
    __syncthreads();
  }

  // ---- heads ----
  const long long offS = (long long)p.Btot;
  const long long offI = (long long)p.Btot * 257;
  const long long offH = (long long)p.Btot * 260;

  { // symbol_pred = h@Ws + bs  -> [B,256]
    v16h a = lds_a(&hTile[w][nl][0], kh);
    for (int t = 0; t < 16; ++t) {
      v16h b = lds_b(&wsT[t * 16 + nl][0], kh);
      v8f c = wmma_f16(a, b, vzero);
      int n = t * 16 + nl;
      #pragma unroll
      for (int g = 0; g < 8; ++g) {
        int m = g + 8 * kh;
        p.out[offS + (base + m) * 256 + n] = c[g] + biasS[n];
      }
    }
  }
  // h -> out
  for (int i = lane; i < 512; i += 32) {
    int m = i >> 5, n = i & 31;
    p.out[offH + (base + m) * 32 + n] = (float)hTile[w][m][n];
  }
  // action / intent (tiny: scalar dots)
  if (lane < 16) {
    float acc = 0.f, i0 = 0.f, i1 = 0.f, i2 = 0.f;
    #pragma unroll
    for (int n = 0; n < 32; ++n) {
      float hv = (float)hTile[w][lane][n];
      acc += hv * WaL[n];
      i0 += hv * WiL[n * 3 + 0];
      i1 += hv * WiL[n * 3 + 1];
      i2 += hv * WiL[n * 3 + 2];
    }
    p.out[base + lane] = fast_tanh(acc + baL[0]);
    long long io = offI + (base + lane) * 3;
    p.out[io + 0] = i0 + biL[0];
    p.out[io + 1] = i1 + biL[1];
    p.out[io + 2] = i2 + biL[2];
  }
}

extern "C" void kernel_launch(void* const* d_in, const int* in_sizes, int n_in,
                              void* d_out, int out_size, void* d_ws, size_t ws_size,
                              hipStream_t stream) {
  (void)n_in; (void)out_size; (void)d_ws; (void)ws_size;
  Params p;
  p.symbol       = (const int*)  d_in[0];
  p.other_hidden = (const float*)d_in[1];
  p.prev_hidden  = (const float*)d_in[2];
  p.emb          = (const float*)d_in[3];
  p.mem_gate     = (const float*)d_in[4];
  p.Wm = (const float*)d_in[5];  p.bm = (const float*)d_in[6];
  p.Wq = (const float*)d_in[7];  p.bq = (const float*)d_in[8];
  p.Wk = (const float*)d_in[9];  p.bk = (const float*)d_in[10];
  p.Wv = (const float*)d_in[11]; p.bv = (const float*)d_in[12];
  p.W1 = (const float*)d_in[13]; p.b1 = (const float*)d_in[14];
  p.W2 = (const float*)d_in[15]; p.b2 = (const float*)d_in[16];
  p.W3 = (const float*)d_in[17]; p.b3 = (const float*)d_in[18];
  p.W4 = (const float*)d_in[19]; p.b4 = (const float*)d_in[20];
  p.Wa = (const float*)d_in[21]; p.ba = (const float*)d_in[22];
  p.Ws = (const float*)d_in[23]; p.bs = (const float*)d_in[24];
  p.Wi = (const float*)d_in[25]; p.bi = (const float*)d_in[26];
  p.out  = (float*)d_out;
  p.Btot = in_sizes[0];
  int blocks = in_sizes[0] / 64; // 64 agents per 128-thread (4-wave) block
  hipLaunchKernelGGL(social_tnn_kernel, dim3(blocks), dim3(128), 0, stream, p);
}
